// HyperEncoder_18107582120274
// MI455X (gfx1250) — matrix operand order repaired and checked by reading, not compiled, not run on hardware
//
#include <hip/hip_runtime.h>

#define NN 100000   // nodes
#define NE 50000    // hyperedges
#define DIM 128

typedef __attribute__((ext_vector_type(16))) __bf16 v16bf;
typedef __attribute__((ext_vector_type(8)))  float  v8f;

__device__ __forceinline__ unsigned short f2bf(float f) {
  union { float f; unsigned int u; } v; v.f = f;
  unsigned int u = v.u;
  u += 0x7FFFu + ((u >> 16) & 1u);   // round-to-nearest-even
  return (unsigned short)(u >> 16);
}

// ---------------- elementwise helpers ----------------
__global__ void k_zero(float* __restrict__ p, int n) {
  int t = blockIdx.x * blockDim.x + threadIdx.x;
  if (t < n) p[t] = 0.f;
}

__global__ void k_degrees(const int* __restrict__ nidx, const int* __restrict__ eidx,
                          float* __restrict__ Dn, float* __restrict__ De, int n_inc) {
  int t = blockIdx.x * blockDim.x + threadIdx.x;
  if (t < n_inc) {
    atomicAdd(&Dn[nidx[t]], 1.0f);
    atomicAdd(&De[eidx[t]], 1.0f);
  }
}

__global__ void k_invert(float* __restrict__ d, int n) {
  int t = blockIdx.x * blockDim.x + threadIdx.x;
  if (t < n) { float v = d[t]; d[t] = (v > 0.f) ? (1.f / v) : 0.f; }
}

__global__ void k_prelu(float* __restrict__ p, int n, const float* __restrict__ alpha) {
  int t = blockIdx.x * blockDim.x + threadIdx.x;
  if (t < n) { float a = *alpha; float v = p[t]; p[t] = (v >= 0.f) ? v : a * v; }
}

__global__ void k_bias_prelu(float* __restrict__ p, const float* __restrict__ bias,
                             const float* __restrict__ alpha, int n) {
  int t = blockIdx.x * blockDim.x + threadIdx.x;
  if (t < n) {
    float a = *alpha;
    float v = p[t] + bias[t & (DIM - 1)];
    p[t] = (v >= 0.f) ? v : a * v;
  }
}

// ---------------- scatter: accum[tidx[i]] += src[gidx[i]] * scale[tidx[i]] ----------------
// one wave (32 lanes) per incidence pair; each lane handles 4 consecutive floats
__global__ __launch_bounds__(256) void k_scatter(const float* __restrict__ src,
                                                 const int* __restrict__ gidx,
                                                 const int* __restrict__ tidx,
                                                 const float* __restrict__ scale,
                                                 float* __restrict__ accum, int n_inc) {
  const int lane = threadIdx.x & 31;
  const int i = blockIdx.x * 8 + (threadIdx.x >> 5);
  if (i >= n_inc) return;
  const int g = gidx[i];
  const int t = tidx[i];
  const float s = scale[t];
  const float4 v = *(const float4*)(src + (size_t)g * DIM + lane * 4);
  float* dst = accum + (size_t)t * DIM + lane * 4;
  atomicAdd(dst + 0, v.x * s);
  atomicAdd(dst + 1, v.y * s);
  atomicAdd(dst + 2, v.z * s);
  atomicAdd(dst + 3, v.w * s);
}

// ---------------- GEMM: C[M,128] = A[M,128] @ W[128,128]  (bf16 WMMA, f32 accum) ----------
// 256 threads = 8 waves, each wave computes 16 rows x 128 cols.
// W staged in LDS in B-fragment order; each wave stages its A tile in A-fragment order.
__global__ __launch_bounds__(256) void k_gemm_bf16(const float* __restrict__ A,
                                                   const float* __restrict__ W,
                                                   float* __restrict__ C, int M) {
  // sW: [nt(8)][kc(4)][lane(32)][16 bf16]  -> 32 KB
  // sA: per wave [kc(4)][lane(32)][16 bf16] -> 4 KB * 8 = 32 KB
  __shared__ unsigned short sW[8 * 4 * 32 * 16];
  __shared__ unsigned short sA[8][4 * 32 * 16];

  const int tid  = threadIdx.x;
  const int lane = tid & 31;
  const int wave = tid >> 5;

  // ---- stage W (f32 -> bf16) into B-fragment layout ----
  // B order used here (contiguous k, matching A): lane = n + 16*kh (kh = k-chunk half at K=16);
  // within lane, vgpr i holds K = kh*16 + 2i, 2i+1.
  {
    const int k  = tid >> 1;
    const int cb = (tid & 1) * 64;
    const float* wr = W + (size_t)k * DIM + cb;
    const int kc   = k >> 5;
    const int kp   = k & 31;
    const int kh   = (kp >> 4) & 1;
    const int ii   = (kp & 15) >> 1;
    const int elem = 2 * ii + (kp & 1);
    for (int c = 0; c < 64; ++c) {
      const int col = cb + c;
      const int nt = col >> 4, np = col & 15;
      const int ln = np + (kh << 4);
      sW[(((nt * 4 + kc) * 32 + ln) << 4) + elem] = f2bf(wr[c]);
    }
  }
  __syncthreads();

  const int row0 = (blockIdx.x * 8 + wave) * 16;
  if (row0 < M) {  // wave-uniform guard: EXEC stays all-1s inside
    // ---- stage this wave's 16x128 A tile into A-fragment layout ----
    // A order used here (contiguous k, matching B): lane = m + 16*kh (kh = bit4 of k within chunk);
    // vgpr i: K = kh*16 + 2i, 2i+1.
    for (int r = 0; r < 16; ++r) {
      const float4 v = *(const float4*)(A + (size_t)(row0 + r) * DIM + lane * 4);
      const float vv[4] = {v.x, v.y, v.z, v.w};
      #pragma unroll
      for (int j = 0; j < 4; ++j) {
        const int k  = lane * 4 + j;
        const int kc = k >> 5;
        const int kp = k & 31;
        const int kh = (kp >> 4) & 1;
        const int ln = r + (kh << 4);
        const int ii = (kp & 15) >> 1;
        const int elem = 2 * ii + (kp & 1);
        sA[wave][((kc * 32 + ln) << 4) + elem] = f2bf(vv[j]);
      }
    }
    // cross-lane LDS staging must be visible before fragment loads (wave-wide DS in order)
    asm volatile("s_wait_dscnt 0" ::: "memory");

    v8f acc[8] = {};
    #pragma unroll
    for (int kc = 0; kc < 4; ++kc) {
      const v16bf a = *(const v16bf*)&sA[wave][(kc * 32 + lane) << 4];
      #pragma unroll
      for (int nt = 0; nt < 8; ++nt) {
        const v16bf b = *(const v16bf*)&sW[((nt * 4 + kc) * 32 + lane) << 4];
        acc[nt] = __builtin_amdgcn_wmma_f32_16x16x32_bf16(
            false, a, false, b, (short)0, acc[nt], false, false);
      }
    }

    // ---- store D tiles: vgpr r -> row = row0 + r + 8*(lane>=16), col = nt*16 + (lane&15)
    const int rbase = row0 + ((lane >> 4) << 3);
    const int cidx  = lane & 15;
    #pragma unroll
    for (int nt = 0; nt < 8; ++nt) {
      #pragma unroll
      for (int r = 0; r < 8; ++r) {
        C[(size_t)(rbase + r) * DIM + nt * 16 + cidx] = acc[nt][r];
      }
    }
  }
}

// ---------------- host-side orchestration ----------------
static inline int cdiv(int a, int b) { return (a + b - 1) / b; }

extern "C" void kernel_launch(void* const* d_in, const int* in_sizes, int n_in,
                              void* d_out, int out_size, void* d_ws, size_t ws_size,
                              hipStream_t stream) {
  const float* x     = (const float*)d_in[0];
  const int*   hidx  = (const int*)  d_in[1];
  const float* Wn2e0 = (const float*)d_in[2];
  const float* We2n0 = (const float*)d_in[3];
  const float* bias0 = (const float*)d_in[4];
  const float* Wn2e1 = (const float*)d_in[5];
  const float* We2n1 = (const float*)d_in[6];
  const float* bias1 = (const float*)d_in[7];
  const float* alpha = (const float*)d_in[8];

  const int n_inc = in_sizes[1] / 2;
  const int* nidx = hidx;           // row 0: node indices
  const int* eidx = hidx + n_inc;   // row 1: edge indices

  float* ws  = (float*)d_ws;
  float* Dn  = ws;                  // NN floats (holds inverse degrees after k_invert)
  float* De  = ws + NN;             // NE floats
  float* buf = ws + 150016;         // NN*DIM floats, reused for xe and xn

  float* n_out = (float*)d_out;                       // NN*DIM
  float* e_out = (float*)d_out + (size_t)NN * DIM;    // NE*DIM

  const int NN_ELEMS = NN * DIM;    // 12,800,000
  const int NE_ELEMS = NE * DIM;    //  6,400,000

  // degrees -> inverse degrees
  k_zero<<<cdiv(NN, 256), 256, 0, stream>>>(Dn, NN);
  k_zero<<<cdiv(NE, 256), 256, 0, stream>>>(De, NE);
  k_degrees<<<cdiv(n_inc, 256), 256, 0, stream>>>(nidx, eidx, Dn, De, n_inc);
  k_invert<<<cdiv(NN, 256), 256, 0, stream>>>(Dn, NN);
  k_invert<<<cdiv(NE, 256), 256, 0, stream>>>(De, NE);

  const int gN = cdiv(NN, 128);   // gemm blocks for M = NN
  const int gE = cdiv(NE, 128);   // gemm blocks for M = NE
  const int gS = cdiv(n_inc, 8);  // scatter blocks (8 incidences per block)

  // ---------------- layer 0 ----------------
  k_gemm_bf16<<<gN, 256, 0, stream>>>(x, Wn2e0, buf, NN);                    // xe = x @ Wn2e0
  k_zero<<<cdiv(NE_ELEMS, 256), 256, 0, stream>>>(e_out, NE_ELEMS);
  k_scatter<<<gS, 256, 0, stream>>>(buf, nidx, eidx, De, e_out, n_inc);      // e = seg_sum
  k_prelu<<<cdiv(NE_ELEMS, 256), 256, 0, stream>>>(e_out, NE_ELEMS, alpha);
  k_gemm_bf16<<<gE, 256, 0, stream>>>(e_out, We2n0, buf, NE);                // xn = e @ We2n0
  k_zero<<<cdiv(NN_ELEMS, 256), 256, 0, stream>>>(n_out, NN_ELEMS);
  k_scatter<<<gS, 256, 0, stream>>>(buf, eidx, nidx, Dn, n_out, n_inc);      // n = seg_sum
  k_bias_prelu<<<cdiv(NN_ELEMS, 256), 256, 0, stream>>>(n_out, bias0, alpha, NN_ELEMS);

  // ---------------- layer 1 ----------------
  k_gemm_bf16<<<gN, 256, 0, stream>>>(n_out, Wn2e1, buf, NN);
  k_zero<<<cdiv(NE_ELEMS, 256), 256, 0, stream>>>(e_out, NE_ELEMS);
  k_scatter<<<gS, 256, 0, stream>>>(buf, nidx, eidx, De, e_out, n_inc);
  k_prelu<<<cdiv(NE_ELEMS, 256), 256, 0, stream>>>(e_out, NE_ELEMS, alpha);  // e (final output)
  k_gemm_bf16<<<gE, 256, 0, stream>>>(e_out, We2n1, buf, NE);
  k_zero<<<cdiv(NN_ELEMS, 256), 256, 0, stream>>>(n_out, NN_ELEMS);
  k_scatter<<<gS, 256, 0, stream>>>(buf, eidx, nidx, Dn, n_out, n_inc);
  k_bias_prelu<<<cdiv(NN_ELEMS, 256), 256, 0, stream>>>(n_out, bias1, alpha, NN_ELEMS); // n (final)
}
